// Encoder_23519240913123
// MI455X (gfx1250) — compile-verified, matrix operand order checked by reading, not run on hardware
//
#include <hip/hip_runtime.h>
#include <hip/hip_bf16.h>

// Problem constants (from the reference)
#define NN   20000      // nodes
#define EE   320000     // edges
#define INF_ 128        // input feature dim
#define HID  64         // hidden dim / per-head dim
#define NH   4          // heads
#define NL   3          // layers
#define NG   128        // graphs

typedef __attribute__((ext_vector_type(16))) __bf16 v16bf;
typedef __attribute__((ext_vector_type(8)))  float  v8f;

__device__ __forceinline__ unsigned short f32_to_bf16(float f) {
  unsigned int x = __float_as_uint(f);
  return (unsigned short)((x + 0x7FFFu + ((x >> 16) & 1u)) >> 16);
}

// ---------------------------------------------------------------------------
// fp32 -> bf16 (round to nearest even) cast, row-major (used for A operands)
// ---------------------------------------------------------------------------
__global__ void cast_f32_bf16(const float* __restrict__ in,
                              unsigned short* __restrict__ out, int n) {
  int i = blockIdx.x * blockDim.x + threadIdx.x;
  if (i < n) out[i] = f32_to_bf16(in[i]);
}

__global__ void fill_f32(float* __restrict__ p, float v, int n) {
  int i = blockIdx.x * blockDim.x + threadIdx.x;
  if (i < n) p[i] = v;
}

// ---------------------------------------------------------------------------
// Pack a row-major fp32 weight matrix B[K,N] into bf16 WMMA-B fragment order:
//   out[ ((nt*(K/32) + kc)*32 + lane)*16 + j ]  =  bf16( B[kk, nt*16 + lane%16] )
// where j=2r+odd and kk = kc*32 + (r<4 ? (lane/16)*8+2r : 16+(lane/16)*8+2(r-4)) + odd
// (CDNA5 ISA 7.12.2 16-bit B-operand striping). Each lane's 16 values become a
// contiguous 32-byte run -> fragment load is two b128 loads.
// ---------------------------------------------------------------------------
__global__ void pack_b_bf16(const float* __restrict__ B,
                            unsigned short* __restrict__ out, int K, int N) {
  int i = blockIdx.x * blockDim.x + threadIdx.x;
  if (i >= K * N) return;
  const int ntile_sz = K * 16;           // elements per 16-column tile
  const int nt   = i / ntile_sz;
  const int rem  = i % ntile_sz;
  const int kc   = rem >> 9;             // / (32 lanes * 16 elems)
  const int rem2 = rem & 511;
  const int lane = rem2 >> 4;
  const int j    = rem2 & 15;
  const int half = lane >> 4, lm = lane & 15;
  const int r = j >> 1, odd = j & 1;
  const int kk = kc * 32 +
                 ((r < 4) ? (half * 8 + 2 * r) : (16 + half * 8 + 2 * (r - 4))) +
                 odd;
  out[i] = f32_to_bf16(B[(size_t)kk * N + nt * 16 + lm]);
}

// ---------------------------------------------------------------------------
// bf16 WMMA GEMM: C[M,N] = A[M,K] @ B[K,N] + bias[N]   (optional ReLU)
//   A  : row-major bf16 [M,K]
//   Bp : fragment-packed bf16 (see pack_b_bf16)
//
// Block = 128 threads = 4 waves covering a 64(M) x 16(N) macro-tile.
// The packed K x 16 B-panel (contiguous K*32 bytes) is memcpy'd to LDS with
// uint4 (b128) transfers; each wave then computes one 16x16 tile:
//   per k-chunk: 2x global_load_b128 (A) + 2x ds_load_b128 (B) + 1x v_wmma.
// Epilogue has an unguarded fast path for full tiles (M=20000 is a multiple
// of 16, so the guarded path is cold).
// ---------------------------------------------------------------------------
template <int K>
__global__ __launch_bounds__(128)
void gemm_bf16_wmma(const unsigned short* __restrict__ A,
                    const unsigned short* __restrict__ Bp,
                    const float* __restrict__ bias,
                    float* __restrict__ C,
                    int M, int N, int relu) {
  __shared__ alignas(16) unsigned short lds_b[K * 16];

  const int tid  = threadIdx.x;
  const int wave = tid >> 5;
  const int lane = tid & 31;
  const int half = lane >> 4;
  const int lm   = lane & 15;
  const int col  = blockIdx.y * 16;
  const int row  = blockIdx.x * 64 + wave * 16;

  // ---- stage packed B panel: contiguous K*32-byte block -> LDS ----
  {
    const uint4* __restrict__ gsrc =
        (const uint4*)(Bp + (size_t)blockIdx.y * K * 16);
    uint4* ldst = (uint4*)lds_b;
#pragma unroll
    for (int i = tid; i < (K * 16) / 8; i += 128) ldst[i] = gsrc[i];
  }
  __syncthreads();

  if (row < M) {
    int m_ld = row + lm;
    if (m_ld >= M) m_ld = M - 1;          // clamp tail-tile loads
    const unsigned short* __restrict__ arow = A + (size_t)m_ld * K;
    // Hint the next macro-tile's A row into cache while WMMAs run.
    __builtin_prefetch(arow + (size_t)64 * K, 0, 0);

    v8f acc = {};
#pragma unroll
    for (int kc = 0; kc < K / 32; ++kc) {
      const int k0 = kc * 32;
      union { v16bf v; uint4 q[2]; } a;
      union { v16bf v; uint4 q[2]; } b;
      // A fragment: two 16-byte runs (8 consecutive K values each)
      a.q[0] = *(const uint4*)(arow + k0 + half * 8);
      a.q[1] = *(const uint4*)(arow + k0 + 16 + half * 8);
      // B fragment: one contiguous 32-byte run per lane from the LDS panel
      const unsigned short* bp = lds_b + ((size_t)(kc * 32 + lane) << 4);
      b.q[0] = *(const uint4*)(bp);
      b.q[1] = *(const uint4*)(bp + 8);
      acc = __builtin_amdgcn_wmma_f32_16x16x32_bf16(
          /*neg_a=*/false, a.v, /*neg_b=*/false, b.v,
          /*c_mod=*/(short)0, acc, /*reuse_a=*/false, /*reuse_b=*/false);
    }

    const int n = col + lm;
    const float bv = bias ? bias[n] : 0.0f;
    float* __restrict__ cp = C + (size_t)(row + 8 * half) * N + n;
    if (row + 16 <= M) {
      // Fast path: full tile, unguarded 8-store clause.
#pragma unroll
      for (int r = 0; r < 8; ++r) {
        float v = acc[r] + bv;
        if (relu) v = fmaxf(v, 0.0f);
        cp[(size_t)r * N] = v;
      }
    } else {
#pragma unroll
      for (int r = 0; r < 8; ++r) {
        const int m = row + r + 8 * half;
        if (m < M) {
          float v = acc[r] + bv;
          if (relu) v = fmaxf(v, 0.0f);
          C[(size_t)m * N + n] = v;
        }
      }
    }
  }
}

// ---------------------------------------------------------------------------
// Edge-wise attention
// ---------------------------------------------------------------------------
__device__ __forceinline__ void atomicMaxF32(float* addr, float val) {
  unsigned int* ai = (unsigned int*)addr;
  unsigned int old = *ai;
  while (__uint_as_float(old) < val) {
    unsigned int assumed = old;
    old = atomicCAS(ai, assumed, __float_as_uint(val));
    if (old == assumed) break;
  }
}

// alpha[e,h] = (q[dst] . k[src]) / sqrt(C); running segment max into amax[dst,h]
__global__ void edge_logits(const float* __restrict__ q,
                            const float* __restrict__ k,
                            const int* __restrict__ src,
                            const int* __restrict__ dst,
                            float* __restrict__ alpha,
                            float* __restrict__ amax) {
  int t = blockIdx.x * blockDim.x + threadIdx.x;
  if (t >= EE * NH) return;
  const int e = t >> 2, h = t & 3;
  const int s = src[e], d = dst[e];
  const float4* __restrict__ qp =
      (const float4*)(q + (size_t)d * (NH * HID) + h * HID);
  const float4* __restrict__ kp =
      (const float4*)(k + (size_t)s * (NH * HID) + h * HID);
  float acc = 0.0f;
#pragma unroll
  for (int c = 0; c < HID / 4; ++c) {
    const float4 a = qp[c], b = kp[c];
    acc = fmaf(a.x, b.x, acc);
    acc = fmaf(a.y, b.y, acc);
    acc = fmaf(a.z, b.z, acc);
    acc = fmaf(a.w, b.w, acc);
  }
  acc *= 0.125f;                 // 1/sqrt(64)
  alpha[t] = acc;
  atomicMaxF32(&amax[(size_t)d * NH + h], acc);
}

// ex = exp(alpha - amax[dst]); denom[dst,h] += ex  (in-place over alpha buffer)
__global__ void edge_exp(float* __restrict__ alpha,
                         const int* __restrict__ dst,
                         const float* __restrict__ amax,
                         float* __restrict__ denom) {
  int t = blockIdx.x * blockDim.x + threadIdx.x;
  if (t >= EE * NH) return;
  const int e = t >> 2, h = t & 3;
  const int d = dst[e];
  float m = amax[(size_t)d * NH + h];
  if (m < -1.0e29f) m = 0.0f;    // mirrors the isfinite() guard
  const float ex = __expf(alpha[t] - m);
  alpha[t] = ex;
  atomicAdd(&denom[(size_t)d * NH + h], ex);
}

// agg[dst,h,c4*4 .. +3] += (ex/denom) * v[src,h,c4*4 .. +3]
// One thread per (edge, head, 4-channel group) -> coalesced float4 reads.
__global__ void edge_message(const float* __restrict__ ex,
                             const float* __restrict__ denom,
                             const float* __restrict__ v,
                             const int* __restrict__ src,
                             const int* __restrict__ dst,
                             float* __restrict__ agg) {
  int t = blockIdx.x * blockDim.x + threadIdx.x;
  if (t >= EE * NH * (HID / 4)) return;
  const int e   = t / (NH * (HID / 4));
  const int rem = t % (NH * (HID / 4));
  const int h   = rem / (HID / 4);
  const int c4  = rem % (HID / 4);
  const int s = src[e], d = dst[e];
  const float a = ex[(size_t)e * NH + h] /
                  fmaxf(denom[(size_t)d * NH + h], 1e-16f);
  const float4 vv =
      *(const float4*)(v + (size_t)s * (NH * HID) + h * HID + c4 * 4);
  float* __restrict__ ap = agg + (size_t)d * (NH * HID) + h * HID + c4 * 4;
  atomicAdd(&ap[0], a * vv.x);
  atomicAdd(&ap[1], a * vv.y);
  atomicAdd(&ap[2], a * vv.z);
  atomicAdd(&ap[3], a * vv.w);
}

// h[n,c] = relu( mean_h agg[n,h,c] + skip[n,c] )   (in place into h)
__global__ void combine(const float* __restrict__ agg,
                        const float* __restrict__ skip,
                        float* __restrict__ h) {
  int t = blockIdx.x * blockDim.x + threadIdx.x;
  if (t >= NN * HID) return;
  const int n = t / HID, c = t % HID;
  const float* __restrict__ ag = agg + (size_t)n * (NH * HID) + c;
  const float s = 0.25f * (ag[0] + ag[HID] + ag[2 * HID] + ag[3 * HID]);
  h[t] = fmaxf(s + skip[t], 0.0f);
}

// out[batch[n], c] += h[n, c]
__global__ void pool(const float* __restrict__ h,
                     const int* __restrict__ batch,
                     float* __restrict__ out) {
  int t = blockIdx.x * blockDim.x + threadIdx.x;
  if (t >= NN * HID) return;
  const int n = t / HID, c = t % HID;
  atomicAdd(&out[(size_t)batch[n] * HID + c], h[t]);
}

// ---------------------------------------------------------------------------
// Host-side orchestration
// ---------------------------------------------------------------------------
static inline int cdiv(int a, int b) { return (a + b - 1) / b; }

extern "C" void kernel_launch(void* const* d_in, const int* in_sizes, int n_in,
                              void* d_out, int out_size, void* d_ws, size_t ws_size,
                              hipStream_t stream) {
  (void)in_sizes; (void)n_in; (void)out_size; (void)ws_size;

  const float* x     = (const float*)d_in[0];
  const int*   eidx  = (const int*)d_in[1];      // [2,E]
  const int*   batch = (const int*)d_in[2];      // [N]
  const float* lin_W = (const float*)d_in[3];    // [IN,HID]
  const float* lin_b = (const float*)d_in[4];    // [HID]
  const float* Wq    = (const float*)d_in[5];    // [L,HID,NH*HID]
  const float* bq    = (const float*)d_in[6];    // [L,NH*HID]
  const float* Wk    = (const float*)d_in[7];
  const float* bk    = (const float*)d_in[8];
  const float* Wv    = (const float*)d_in[9];
  const float* bv    = (const float*)d_in[10];
  const float* Ws    = (const float*)d_in[11];   // [L,HID,HID]
  const float* bs    = (const float*)d_in[12];   // [L,HID]
  float* out = (float*)d_out;                    // [G,HID]

  const int* src = eidx;
  const int* dst = eidx + EE;

  // -------- carve workspace --------
  char* w = (char*)d_ws;
  auto carve = [&](size_t bytes) -> char* {
    char* p = w;
    w += (bytes + 255) & ~(size_t)255;
    return p;
  };
  unsigned short* xb  = (unsigned short*)carve((size_t)NN * INF_ * 2);
  unsigned short* lwb = (unsigned short*)carve((size_t)INF_ * HID * 2);
  unsigned short* wqb = (unsigned short*)carve((size_t)NL * HID * NH * HID * 2);
  unsigned short* wkb = (unsigned short*)carve((size_t)NL * HID * NH * HID * 2);
  unsigned short* wvb = (unsigned short*)carve((size_t)NL * HID * NH * HID * 2);
  unsigned short* wsb = (unsigned short*)carve((size_t)NL * HID * HID * 2);
  float* h    = (float*)carve((size_t)NN * HID * 4);
  unsigned short* hb = (unsigned short*)carve((size_t)NN * HID * 2);
  float* q    = (float*)carve((size_t)NN * NH * HID * 4);
  float* kbuf = (float*)carve((size_t)NN * NH * HID * 4);
  float* vbuf = (float*)carve((size_t)NN * NH * HID * 4);
  float* skip = (float*)carve((size_t)NN * HID * 4);
  float* alp  = (float*)carve((size_t)EE * NH * 4);
  float* amax = (float*)carve((size_t)NN * NH * 4);
  float* den  = (float*)carve((size_t)NN * NH * 4);
  float* agg  = (float*)carve((size_t)NN * NH * HID * 4);

  const int TB = 256;
  auto launch_cast = [&](const float* in, unsigned short* o, int n) {
    cast_f32_bf16<<<cdiv(n, TB), TB, 0, stream>>>(in, o, n);
  };
  auto launch_pack = [&](const float* B, unsigned short* o, int K, int N) {
    pack_b_bf16<<<cdiv(K * N, TB), TB, 0, stream>>>(B, o, K, N);
  };

  // -------- one-time casts / weight packing --------
  launch_cast(x, xb, NN * INF_);
  launch_pack(lin_W, lwb, INF_, HID);
  for (int l = 0; l < NL; ++l) {
    launch_pack(Wq + (size_t)l * HID * NH * HID,
                wqb + (size_t)l * HID * NH * HID, HID, NH * HID);
    launch_pack(Wk + (size_t)l * HID * NH * HID,
                wkb + (size_t)l * HID * NH * HID, HID, NH * HID);
    launch_pack(Wv + (size_t)l * HID * NH * HID,
                wvb + (size_t)l * HID * NH * HID, HID, NH * HID);
    launch_pack(Ws + (size_t)l * HID * HID,
                wsb + (size_t)l * HID * HID, HID, HID);
  }

  const int mb = cdiv(NN, 64);   // 64-row macro-tiles (4 waves/block)

  // -------- input linear: h = x @ lin_W + lin_b --------
  gemm_bf16_wmma<INF_><<<dim3(mb, HID / 16), 128, 0, stream>>>(
      xb, lwb, lin_b, h, NN, HID, /*relu=*/0);

  // -------- transformer layers --------
  for (int l = 0; l < NL; ++l) {
    launch_cast(h, hb, NN * HID);

    const unsigned short* wq_l = wqb + (size_t)l * HID * NH * HID;
    const unsigned short* wk_l = wkb + (size_t)l * HID * NH * HID;
    const unsigned short* wv_l = wvb + (size_t)l * HID * NH * HID;
    const unsigned short* ws_l = wsb + (size_t)l * HID * HID;

    gemm_bf16_wmma<HID><<<dim3(mb, (NH * HID) / 16), 128, 0, stream>>>(
        hb, wq_l, bq + (size_t)l * NH * HID, q, NN, NH * HID, 0);
    gemm_bf16_wmma<HID><<<dim3(mb, (NH * HID) / 16), 128, 0, stream>>>(
        hb, wk_l, bk + (size_t)l * NH * HID, kbuf, NN, NH * HID, 0);
    gemm_bf16_wmma<HID><<<dim3(mb, (NH * HID) / 16), 128, 0, stream>>>(
        hb, wv_l, bv + (size_t)l * NH * HID, vbuf, NN, NH * HID, 0);
    gemm_bf16_wmma<HID><<<dim3(mb, HID / 16), 128, 0, stream>>>(
        hb, ws_l, bs + (size_t)l * HID, skip, NN, HID, 0);

    fill_f32<<<cdiv(NN * NH, TB), TB, 0, stream>>>(amax, -3.0e38f, NN * NH);
    fill_f32<<<cdiv(NN * NH, TB), TB, 0, stream>>>(den, 0.0f, NN * NH);
    fill_f32<<<cdiv(NN * NH * HID, TB), TB, 0, stream>>>(agg, 0.0f, NN * NH * HID);

    edge_logits<<<cdiv(EE * NH, TB), TB, 0, stream>>>(q, kbuf, src, dst, alp, amax);
    edge_exp<<<cdiv(EE * NH, TB), TB, 0, stream>>>(alp, dst, amax, den);
    edge_message<<<cdiv(EE * NH * (HID / 4), TB), TB, 0, stream>>>(
        alp, den, vbuf, src, dst, agg);
    combine<<<cdiv(NN * HID, TB), TB, 0, stream>>>(agg, skip, h);
  }

  // -------- global add pool --------
  fill_f32<<<cdiv(NG * HID, TB), TB, 0, stream>>>(out, 0.0f, NG * HID);
  pool<<<cdiv(NN * HID, TB), TB, 0, stream>>>(h, batch, out);
}
